// OnlineSoftBandGRUSFC2D_85950885528366
// MI455X (gfx1250) — matrix operand order repaired and checked by reading, not compiled, not run on hardware
//
#include <hip/hip_runtime.h>

// ---------------------------------------------------------------------------
// OnlineSoftBandGRUSFC2D for MI455X (gfx1250, wave32, WMMA)
//
// Persistent pipeline: 12 workgroups (1 per layer / WGP) stream T=1000 frames
// through a ring buffer in global scratch, synchronized with agent-scope
// acquire/release atomics.  Per cell the two 288x96x512 GEMMs run on
// v_wmma_f32_16x16x32_f16 with all six gate tiles fused in registers.
// ---------------------------------------------------------------------------

typedef __attribute__((ext_vector_type(16))) _Float16 v16h;
typedef __attribute__((ext_vector_type(8)))  _Float16 v8h;
typedef __attribute__((ext_vector_type(8)))  float    v8f;

#define BB 8
#define CCH 96
#define TTT 1000
#define FFR 64
#define LLAY 12
#define KW 3
#define C3 (3 * CCH)        /* 288 */
#define NCOL (BB * FFR)     /* 512 columns = (b,f) pairs */
#define ELEMS (CCH * NCOL)  /* 49152 per frame */
#define RDEPTH 4            /* ring depth (frames) per layer interface */
#define NTHREADS 512        /* 16 wave32 */
#define NWAVES 16
#define EPSV 1e-6f

// d_ws layout (bytes)
#define OFF_FLAGS ((size_t)0)                                     /* produced[16], consumed[16] */
#define OFF_HST   ((size_t)256)                                   /* L*ELEMS f32 hidden state   */
#define OFF_SBUF  (OFF_HST  + (size_t)LLAY * ELEMS * 4)           /* L*ELEMS f32 (h_new + y)    */
#define OFF_RING  (OFF_SBUF + (size_t)LLAY * ELEMS * 4)           /* (L-1)*RD*ELEMS f32         */
#define OFF_W16   (OFF_RING + (size_t)(LLAY - 1) * RDEPTH * ELEMS * 4) /* L * 2*288*96 f16      */
#define WS_NEED   (OFF_W16  + (size_t)LLAY * 2 * C3 * CCH * 2)    /* ~14.7 MB total             */

#define SMEM_BYTES (3 * ELEMS * 2 + 1248 * 4)                     /* ~293 KB LDS */

// A-fragment (16x32 f16) loader: lane m=lane&15,
// half-lanes select K groups {0..15} / {16..31} (matching the B fragment); each group is a
// contiguous 32B run in a row-major f16 weight matrix -> two b128 loads.
__device__ __forceinline__ v16h load_a_tile(const _Float16* W, int row, int kbase) {
  const v8h* p = (const v8h*)(W + (size_t)row * CCH + kbase);
  v8h lo = p[0];
  v8h hi = p[1];
  v16h a;
#pragma unroll
  for (int h = 0; h < 8; ++h) { a[h] = lo[h]; a[h + 8] = hi[h]; }
  return a;
}

__global__ void obg_init_kernel(float* wsf, unsigned long long n) {
  unsigned long long i = (unsigned long long)blockIdx.x * blockDim.x + threadIdx.x;
  if (i < n) wsf[i] = 0.0f;
}

__global__ __launch_bounds__(NTHREADS)
void obg_pipeline_kernel(const float* __restrict__ x,
                         const float* __restrict__ inw,  const float* __restrict__ hnw,
                         const float* __restrict__ xw,   const float* __restrict__ xb,
                         const float* __restrict__ hmw,  const float* __restrict__ hmb,
                         const float* __restrict__ pw,   const float* __restrict__ pb,
                         const float* __restrict__ onw,
                         float* __restrict__ out, char* __restrict__ ws) {
  const int l    = blockIdx.x;           // one layer per workgroup (per WGP)
  const int tid  = threadIdx.x;
  const int lane = tid & 31;
  const int wave = tid >> 5;

  extern __shared__ char smem[];
  _Float16* sYnT = (_Float16*)smem;             // [NCOL][CCH] normalized input (B-op layout)
  _Float16* sHn  = sYnT + ELEMS;                // [CCH][NCOL] rms(h) (conv source)
  _Float16* sHcT = sHn  + ELEMS;                // [NCOL][CCH] dwconv result (B-op layout)
  float* sP    = (float*)(sHcT + ELEMS);
  float* s_inw = sP;            // 96
  float* s_hnw = sP + 96;       // 96
  float* s_onw = sP + 192;      // 96
  float* s_hmb = sP + 288;      // 96
  float* s_hmw = sP + 384;      // 288
  float* s_xb  = sP + 672;      // 288
  float* s_pb  = sP + 960;      // 288

  int*   produced = (int*)(ws + OFF_FLAGS);
  int*   consumed = produced + 16;
  float* hst_l    = (float*)(ws + OFF_HST)  + (size_t)l * ELEMS;
  float* sbuf_l   = (float*)(ws + OFF_SBUF) + (size_t)l * ELEMS;
  float* ring     = (float*)(ws + OFF_RING);
  const float* ringIn = (l > 0)       ? ring + (size_t)(l - 1) * RDEPTH * ELEMS : ring;
  float* ringOut      = (l < LLAY-1)  ? ring + (size_t)l * RDEPTH * ELEMS       : ring;
  _Float16* xw16 = (_Float16*)(ws + OFF_W16) + (size_t)l * 2 * C3 * CCH;
  _Float16* pw16 = xw16 + (size_t)C3 * CCH;

  // ---- prologue: convert this layer's GEMM weights to f16, stage small params
  {
    const float* xwl = xw + (size_t)l * C3 * CCH;
    const float* pwl = pw + (size_t)l * C3 * CCH;
    for (int i = tid; i < C3 * CCH; i += NTHREADS) {
      xw16[i] = (_Float16)xwl[i];
      pw16[i] = (_Float16)pwl[i];
    }
    for (int i = tid; i < CCH; i += NTHREADS) {
      s_inw[i] = inw[l * CCH + i];  s_hnw[i] = hnw[l * CCH + i];
      s_onw[i] = onw[l * CCH + i];  s_hmb[i] = hmb[l * CCH + i];
    }
    for (int i = tid; i < CCH * KW; i += NTHREADS) s_hmw[i] = hmw[(size_t)l * CCH * KW + i];
    for (int i = tid; i < C3; i += NTHREADS) { s_xb[i] = xb[l * C3 + i]; s_pb[i] = pb[l * C3 + i]; }
    __threadfence();
    __syncthreads();
  }

  const int j  = tid;            // this thread's column for the elementwise phases
  const int bj = j >> 6;
  const int fj = j & 63;

  for (int t = 0; t < TTT; ++t) {
    // ---- wait until the previous layer has produced frame t
    if (l > 0 && tid == 0) {
      while (__hip_atomic_load(&produced[l - 1], __ATOMIC_ACQUIRE, __HIP_MEMORY_SCOPE_AGENT) <= t)
        __builtin_amdgcn_s_sleep(2);
    }
    __syncthreads();

    // ---- Phase A: in_norm(y) -> sYnT (f16, [col][k] so B-fragments are contiguous)
    {
      float ss = 0.f;
      if (l == 0) {
        const float* base = x + (((size_t)bj * CCH) * TTT + t) * FFR + fj;
        for (int c = 0; c < CCH; ++c) { float v = base[(size_t)c * TTT * FFR]; ss += v * v; }
        float sc = __frsqrt_rn(ss * (1.0f / CCH) + EPSV);
        for (int c = 0; c < CCH; ++c) {
          float v = base[(size_t)c * TTT * FFR];
          sYnT[j * CCH + c] = (_Float16)(v * sc * s_inw[c]);
        }
        if (t + 1 < TTT) __builtin_prefetch(base + FFR, 0, 1);  // next frame of x
      } else {
        const float* base = ringIn + (size_t)(t % RDEPTH) * ELEMS + j;
        for (int c = 0; c < CCH; ++c) { float v = base[c * NCOL]; ss += v * v; }
        float sc = __frsqrt_rn(ss * (1.0f / CCH) + EPSV);
        for (int c = 0; c < CCH; ++c) {
          float v = base[c * NCOL];
          sYnT[j * CCH + c] = (_Float16)(v * sc * s_inw[c]);
        }
      }
    }

    // ---- Phase B: hid_norm(h_prev) -> sHn (f16, [k][col] for conv neighbors)
    {
      float ss = 0.f;
      const float* hb = hst_l + j;
      for (int c = 0; c < CCH; ++c) { float v = hb[c * NCOL]; ss += v * v; }
      float sc = __frsqrt_rn(ss * (1.0f / CCH) + EPSV);
      for (int c = 0; c < CCH; ++c) {
        float v = hb[c * NCOL];
        sHn[c * NCOL + j] = (_Float16)(v * sc * s_hnw[c]);
      }
    }
    __syncthreads();

    // ---- depthwise conv (K=3 along f) -> sHcT
    for (int c = 0; c < CCH; ++c) {
      float mid = (float)sHn[c * NCOL + j];
      float lft = (fj > 0)       ? (float)sHn[c * NCOL + j - 1] : 0.f;
      float rgt = (fj < FFR - 1) ? (float)sHn[c * NCOL + j + 1] : 0.f;
      sHcT[j * CCH + c] =
          (_Float16)(lft * s_hmw[c * KW] + mid * s_hmw[c * KW + 1] + rgt * s_hmw[c * KW + 2] + s_hmb[c]);
    }
    __syncthreads();

    // ---- Phase C: WMMA GEMMs + fused GRU gating
    {
      const int n    = lane & 15;          // output column within tile
      const int kb16 = (lane >> 4) * 16;   // B fragment K base per half-wave
      const int klo8 = (lane >> 4) * 16;   // A fragment K base per half-wave
      const int m    = lane & 15;          // A row within tile
      const int mo   = (lane >> 4) * 8;    // accumulator row offset
      for (int job = wave; job < 192; job += NWAVES) {     // 6 M-tiles x 32 N-tiles
        const int cm = (job >> 5) * 16;    // channel-tile base (0..80)
        const int jt = (job & 31) * 16;    // column-tile base (0..496)
        v8f aXR = {}, aXZ = {}, aXN = {}, aHR = {}, aHZ = {}, aHN = {};
#pragma unroll
        for (int kk = 0; kk < 3; ++kk) {   // K = 96 = 3 x 32
          const int k0 = kk * 32;
          v16h bX = *(const v16h*)(sYnT + (size_t)(jt + n) * CCH + k0 + kb16);
          v16h bH = *(const v16h*)(sHcT + (size_t)(jt + n) * CCH + k0 + kb16);
          aXR = __builtin_amdgcn_wmma_f32_16x16x32_f16(false, load_a_tile(xw16, cm + m,          k0 + klo8), false, bX, (short)0, aXR, false, false);
          aXZ = __builtin_amdgcn_wmma_f32_16x16x32_f16(false, load_a_tile(xw16, cm + CCH + m,    k0 + klo8), false, bX, (short)0, aXZ, false, false);
          aXN = __builtin_amdgcn_wmma_f32_16x16x32_f16(false, load_a_tile(xw16, cm + 2*CCH + m,  k0 + klo8), false, bX, (short)0, aXN, false, false);
          aHR = __builtin_amdgcn_wmma_f32_16x16x32_f16(false, load_a_tile(pw16, cm + m,          k0 + klo8), false, bH, (short)0, aHR, false, false);
          aHZ = __builtin_amdgcn_wmma_f32_16x16x32_f16(false, load_a_tile(pw16, cm + CCH + m,    k0 + klo8), false, bH, (short)0, aHZ, false, false);
          aHN = __builtin_amdgcn_wmma_f32_16x16x32_f16(false, load_a_tile(pw16, cm + 2*CCH + m,  k0 + klo8), false, bH, (short)0, aHN, false, false);
        }
        // GRU gating fused in registers; s = h_new + y -> sbuf (global, L2-hot)
        const int jj  = jt + n;
        const int bb2 = jj >> 6, ff2 = jj & 63;
#pragma unroll
        for (int r = 0; r < 8; ++r) {
          const int c = cm + mo + r;
          float xr = aXR[r] + s_xb[c];           float hr = aHR[r] + s_pb[c];
          float xz = aXZ[r] + s_xb[c + CCH];     float hz = aHZ[r] + s_pb[c + CCH];
          float xn = aXN[r] + s_xb[c + 2*CCH];   float hn = aHN[r] + s_pb[c + 2*CCH];
          float rst = 1.f / (1.f + __expf(-(xr + hr)));
          float upd = 1.f / (1.f + __expf(-(xz + hz)));
          float e2  = __expf(2.f * (xn + rst * hn));
          float cand = (e2 - 1.f) / (e2 + 1.f);
          float hp = hst_l[c * NCOL + jj];
          float yv = (l == 0)
              ? x[(((size_t)bb2 * CCH + c) * TTT + t) * FFR + ff2]
              : ringIn[(size_t)(t % RDEPTH) * ELEMS + c * NCOL + jj];
          sbuf_l[c * NCOL + jj] = (1.f - upd) * cand + upd * hp + yv;
        }
      }
    }
    __threadfence();
    __syncthreads();

    // frame t of the input ring fully consumed (Phase A + gating reads done)
    if (tid == 0 && l > 0)
      __hip_atomic_store(&consumed[l - 1], t + 1, __ATOMIC_RELEASE, __HIP_MEMORY_SCOPE_AGENT);
    // backpressure: don't overwrite ring slot t%RD until consumer is past t-RD
    if (tid == 0 && l < LLAY - 1 && t >= RDEPTH) {
      while (__hip_atomic_load(&consumed[l], __ATOMIC_ACQUIRE, __HIP_MEMORY_SCOPE_AGENT) < t - RDEPTH + 1)
        __builtin_amdgcn_s_sleep(2);
    }
    __syncthreads();

    // ---- Phase D: out_norm(h_new + y) -> hidden state + next layer / d_out
    {
      float ss = 0.f;
      const float* sb = sbuf_l + j;
      for (int c = 0; c < CCH; ++c) { float v = sb[c * NCOL]; ss += v * v; }
      float sc = __frsqrt_rn(ss * (1.0f / CCH) + EPSV);
      float* ho = hst_l + j;
      if (l < LLAY - 1) {
        float* ro = ringOut + (size_t)(t % RDEPTH) * ELEMS + j;
        for (int c = 0; c < CCH; ++c) {
          float o = sb[c * NCOL] * sc * s_onw[c];
          ho[c * NCOL] = o;  ro[c * NCOL] = o;
        }
      } else {
        float* oo = out + (((size_t)bj * CCH) * TTT + t) * FFR + fj;
        for (int c = 0; c < CCH; ++c) {
          float o = sb[c * NCOL] * sc * s_onw[c];
          ho[c * NCOL] = o;  oo[(size_t)c * TTT * FFR] = o;
        }
      }
    }
    __threadfence();
    __syncthreads();
    if (tid == 0 && l < LLAY - 1)
      __hip_atomic_store(&produced[l], t + 1, __ATOMIC_RELEASE, __HIP_MEMORY_SCOPE_AGENT);
  }
}

extern "C" void kernel_launch(void* const* d_in, const int* in_sizes, int n_in,
                              void* d_out, int out_size, void* d_ws, size_t ws_size,
                              hipStream_t stream) {
  (void)in_sizes; (void)n_in; (void)out_size;
  if (ws_size < WS_NEED) return;  // needs ~14.7 MB of scratch

  const float* x   = (const float*)d_in[0];
  const float* inw = (const float*)d_in[1];
  const float* hnw = (const float*)d_in[2];
  const float* xw  = (const float*)d_in[3];
  const float* xb  = (const float*)d_in[4];
  const float* hmw = (const float*)d_in[5];
  const float* hmb = (const float*)d_in[6];
  const float* pw  = (const float*)d_in[7];
  const float* pb  = (const float*)d_in[8];
  const float* onw = (const float*)d_in[9];
  float* out = (float*)d_out;
  char*  ws  = (char*)d_ws;

  // zero flags + hidden states every launch (graph-replay safe, no statics)
  unsigned long long zn = (256ull + (unsigned long long)LLAY * ELEMS * 4ull) / 4ull;
  hipLaunchKernelGGL(obg_init_kernel, dim3((unsigned)((zn + 255) / 256)), dim3(256), 0, stream,
                     (float*)ws, zn);

  (void)hipFuncSetAttribute((const void*)obg_pipeline_kernel,
                            hipFuncAttributeMaxDynamicSharedMemorySize, SMEM_BYTES);
  hipLaunchKernelGGL(obg_pipeline_kernel, dim3(LLAY), dim3(NTHREADS), SMEM_BYTES, stream,
                     x, inw, hnw, xw, xb, hmw, hmb, pw, pb, onw, out, ws);
}
